// PredictorRNN_29764123361825
// MI455X (gfx1250) — compile-verified, matrix operand order checked by reading, not run on hardware
//
#include <hip/hip_runtime.h>

// ---------------- problem constants ----------------
#define B_    256
#define T_    256
#define I_    64
#define H_    512
#define HOR   8
#define MAXT  (T_ + HOR)    // 264
#define NWG   16            // persistent workgroups; each owns 32 columns of H

typedef __attribute__((ext_vector_type(16))) __bf16 v16bf;
typedef __attribute__((ext_vector_type(8)))  __bf16 v8bf;
typedef __attribute__((ext_vector_type(8)))  float  v8f;

// ---------------- workspace byte offsets ----------------
static const size_t OFF_WHH = 0;                                   // 1536x512 bf16
static const size_t OFF_WIH = OFF_WHH + (size_t)1536 * 512 * 2;    // 1536x64  bf16
static const size_t OFF_FCW = OFF_WIH + (size_t)1536 * 64 * 2;     // 64x512   bf16
static const size_t OFF_SEQ = OFF_FCW + (size_t)64 * 512 * 2;      // 256x264x64 bf16
static const size_t OFF_HBF = OFF_SEQ + (size_t)B_ * MAXT * I_ * 2;// 2 x 256x512 bf16 (double buffer)
static const size_t OFF_BAR = OFF_HBF + (size_t)2 * B_ * H_ * 2;   // barrier counter

// ---------------- LDS layout (u16 units) ----------------
#define S_WHH 0        // 96 rows x 512 (this WG's slice of w_hh, gate-major: r,z,n x 32 cols)
#define S_WIH 49152    // 96 rows x 64  (slice of w_ih)
#define S_F32 55296    // float area: bih[96], bhh[96], h_slice f32 [256][32]
#define SMEM_BYTES (110592 + (192 + 256 * 32) * 4)   // = 144128 bytes

__device__ __forceinline__ unsigned short f2bf(float f) {
  unsigned u = __float_as_uint(f);
  unsigned r = u + 0x7FFFu + ((u >> 16) & 1u);   // round-to-nearest-even
  return (unsigned short)(r >> 16);
}

// CDNA5 16-bit fragment: per lane two contiguous 16B runs (V0-3 = K0..7|K8..15 by
// lane[4], V4-7 = K16..23|K24..31). Works for both A (row=lane&15) and B (col=lane&15).
__device__ __forceinline__ v16bf ld_frag(const unsigned short* p) {
  v8bf lo = *(const v8bf*)p;
  v8bf hi = *(const v8bf*)(p + 16);
  return __builtin_shufflevector(lo, hi, 0,1,2,3,4,5,6,7,8,9,10,11,12,13,14,15);
}

__device__ __forceinline__ v8f wmma_bf16(v16bf a, v16bf b, v8f c) {
  return __builtin_amdgcn_wmma_f32_16x16x32_bf16(false, a, false, b, (short)0, c,
                                                 false, false);
}

__device__ __forceinline__ void grid_barrier(unsigned* cnt, unsigned target) {
  __syncthreads();
  if (threadIdx.x == 0) {
    __threadfence();
    atomicAdd(cnt, 1u);
    while (__hip_atomic_load(cnt, __ATOMIC_RELAXED, __HIP_MEMORY_SCOPE_AGENT) < target)
      __builtin_amdgcn_s_sleep(1);
    __threadfence();
  }
  __syncthreads();
}

// ---------------- init: f32 -> bf16 conversions, zero h / barrier ----------------
__global__ void gru_init_convert(const float* __restrict__ x,
                                 const float* __restrict__ wih,
                                 const float* __restrict__ whh,
                                 const float* __restrict__ fcw,
                                 unsigned short* __restrict__ whh_bf,
                                 unsigned short* __restrict__ wih_bf,
                                 unsigned short* __restrict__ fcw_bf,
                                 unsigned short* __restrict__ seq_bf,
                                 unsigned short* __restrict__ h_bf,
                                 unsigned* __restrict__ bar) {
  size_t i = (size_t)blockIdx.x * blockDim.x + threadIdx.x;
  size_t stride = (size_t)gridDim.x * blockDim.x;
  if (i == 0) bar[0] = 0u;
  for (size_t k = i; k < (size_t)1536 * 512; k += stride) whh_bf[k] = f2bf(whh[k]);
  for (size_t k = i; k < (size_t)1536 * 64;  k += stride) wih_bf[k] = f2bf(wih[k]);
  for (size_t k = i; k < (size_t)64 * 512;   k += stride) fcw_bf[k] = f2bf(fcw[k]);
  for (size_t k = i; k < (size_t)B_ * T_ * I_; k += stride) {
    size_t b = k / (T_ * I_), rem = k % (T_ * I_);
    size_t t = rem / I_,      q   = rem % I_;
    seq_bf[(b * MAXT + t) * I_ + q] = f2bf(x[k]);
  }
  for (size_t k = i; k < (size_t)B_ * H_; k += stride) h_bf[k] = 0;  // h buffer 0 = zeros
}

// ---------------- persistent GRU rollout ----------------
__global__ __launch_bounds__(256, 1)
void gru_persistent(const float* __restrict__ b_ih, const float* __restrict__ b_hh,
                    const float* __restrict__ fc_b,
                    const unsigned short* __restrict__ whh_bf,
                    const unsigned short* __restrict__ wih_bf,
                    const unsigned short* __restrict__ fcw_bf,
                    unsigned short* __restrict__ seq_bf,
                    unsigned short* __restrict__ h_bf,
                    float* __restrict__ out,
                    unsigned* __restrict__ bar) {
  extern __shared__ unsigned short smem_u16[];
  const int tid  = threadIdx.x;
  const int wgid = blockIdx.x;
  const int col0 = wgid * 32;          // this WG owns h columns [col0, col0+32)

  float* fsm   = (float*)&smem_u16[S_F32];
  float* bih_l = fsm;                  // [96]  gate-major biases (r,z,n x 32)
  float* bhh_l = fsm + 96;             // [96]
  float* hsl   = fsm + 192;            // [256][32] f32 h slice (persistent)

  // ---- stage weight slices into LDS once; they stay resident for all 2076 steps ----
  for (int i = tid; i < 96 * 64; i += 256) {       // w_hh slice: 96 rows x 512 u16, 16B chunks
    int lr = i >> 6, part = i & 63;
    int grow = ((lr >> 5) * H_) + col0 + (lr & 31);
    *(uint4*)&smem_u16[(size_t)lr * 512 + part * 8] =
        *(const uint4*)(whh_bf + (size_t)grow * 512 + part * 8);
  }
  for (int i = tid; i < 96 * 8; i += 256) {        // w_ih slice: 96 rows x 64 u16
    int lr = i >> 3, part = i & 7;
    int grow = ((lr >> 5) * H_) + col0 + (lr & 31);
    *(uint4*)&smem_u16[S_WIH + (size_t)lr * 64 + part * 8] =
        *(const uint4*)(wih_bf + (size_t)grow * 64 + part * 8);
  }
  if (tid < 96) {
    int grow = ((tid >> 5) * H_) + col0 + (tid & 31);
    bih_l[tid] = b_ih[grow];
    bhh_l[tid] = b_hh[grow];
  }
  for (int i = tid; i < 256 * 32; i += 256) hsl[i] = 0.0f;
  __syncthreads();

  const int wv   = tid >> 5;           // wave 0..7
  const int lane = tid & 31;
  const int lm   = lane & 15;
  const int hb   = lane >> 4;
  const int r0   = (wv * 2 + 0) * 16 + lm;   // A rows for this wave's two B-tiles
  const int r1   = (wv * 2 + 1) * 16 + lm;

  unsigned ep = 0;
  int cur = 0;

  for (int p = 0; p < HOR; ++p) {
    const int Tlen = T_ + p;
    for (int t = 0; t < Tlen; ++t) {
      v8f aR[2][2] = {}, aZ[2][2] = {}, aHN[2][2] = {}, aXN[2][2] = {};

      // ---- x-projection: K = 64 (2 k-chunks), A = seq[:,t,:], B = w_ih slice (LDS) ----
      const unsigned short* x0 = seq_bf + ((size_t)r0 * MAXT + t) * I_;
      const unsigned short* x1 = seq_bf + ((size_t)r1 * MAXT + t) * I_;
#pragma unroll
      for (int kc = 0; kc < 2; ++kc) {
        const int ko = kc * 32 + hb * 8;
        v16bf a0 = ld_frag(x0 + ko);
        v16bf a1 = ld_frag(x1 + ko);
#pragma unroll
        for (int c = 0; c < 2; ++c) {
          const int nl = c * 16 + lm;
          v16bf br = ld_frag(&smem_u16[S_WIH + (size_t)(nl)      * 64 + ko]);
          v16bf bz = ld_frag(&smem_u16[S_WIH + (size_t)(32 + nl) * 64 + ko]);
          v16bf bn = ld_frag(&smem_u16[S_WIH + (size_t)(64 + nl) * 64 + ko]);
          aR[0][c]  = wmma_bf16(a0, br, aR[0][c]);  aR[1][c]  = wmma_bf16(a1, br, aR[1][c]);
          aZ[0][c]  = wmma_bf16(a0, bz, aZ[0][c]);  aZ[1][c]  = wmma_bf16(a1, bz, aZ[1][c]);
          aXN[0][c] = wmma_bf16(a0, bn, aXN[0][c]); aXN[1][c] = wmma_bf16(a1, bn, aXN[1][c]);
        }
      }

      // ---- recurrence: K = 512 (16 k-chunks), A = h (global bf16, L2), B = w_hh (LDS) ----
      const unsigned short* hc = h_bf + (size_t)cur * (B_ * H_);
      const unsigned short* h0 = hc + (size_t)r0 * H_;
      const unsigned short* h1 = hc + (size_t)r1 * H_;
      for (int kc = 0; kc < 16; ++kc) {
        const int ko = kc * 32 + hb * 8;
        if (kc < 15) {                       // -> global_prefetch_b8 for next k-chunk
          __builtin_prefetch(h0 + ko + 32, 0, 1);
          __builtin_prefetch(h1 + ko + 32, 0, 1);
        }
        v16bf a0 = ld_frag(h0 + ko);
        v16bf a1 = ld_frag(h1 + ko);
#pragma unroll
        for (int c = 0; c < 2; ++c) {
          const int nl = c * 16 + lm;
          v16bf br = ld_frag(&smem_u16[(size_t)(nl)      * 512 + ko]);
          v16bf bz = ld_frag(&smem_u16[(size_t)(32 + nl) * 512 + ko]);
          v16bf bn = ld_frag(&smem_u16[(size_t)(64 + nl) * 512 + ko]);
          aR[0][c]  = wmma_bf16(a0, br, aR[0][c]);  aR[1][c]  = wmma_bf16(a1, br, aR[1][c]);
          aZ[0][c]  = wmma_bf16(a0, bz, aZ[0][c]);  aZ[1][c]  = wmma_bf16(a1, bz, aZ[1][c]);
          aHN[0][c] = wmma_bf16(a0, bn, aHN[0][c]); aHN[1][c] = wmma_bf16(a1, bn, aHN[1][c]);
        }
      }

      // ---- gate elementwise + h update (PyTorch order r,z,n) ----
      unsigned short* hn_out = h_bf + (size_t)(cur ^ 1) * (B_ * H_);
#pragma unroll
      for (int bi = 0; bi < 2; ++bi) {
        const int btr = (wv * 2 + bi) * 16;
#pragma unroll
        for (int c = 0; c < 2; ++c) {
          const int nl = c * 16 + lm;
          const float brb = bih_l[nl]      + bhh_l[nl];
          const float bzb = bih_l[32 + nl] + bhh_l[32 + nl];
          const float bxn = bih_l[64 + nl];
          const float bhn = bhh_l[64 + nl];
#pragma unroll
          for (int vi = 0; vi < 8; ++vi) {
            const int b = btr + hb * 8 + vi;                 // C/D layout: M = vi + 8*(lane>>4)
            const float r = 1.0f / (1.0f + __expf(-(aR[bi][c][vi] + brb)));
            const float z = 1.0f / (1.0f + __expf(-(aZ[bi][c][vi] + bzb)));
            const float nn = tanhf(aXN[bi][c][vi] + bxn + r * (aHN[bi][c][vi] + bhn));
            const float ho = hsl[b * 32 + nl];
            const float hv = (1.0f - z) * nn + z * ho;
            hsl[b * 32 + nl] = hv;
            hn_out[(size_t)b * H_ + col0 + nl] = f2bf(hv);
          }
        }
      }

      ++ep;
      grid_barrier(bar, ep * NWG);
      cur ^= 1;
    }

    // ---- fc head: pred = h_final @ fc_w.T + fc_b; append to sequence ----
    if (wv < 4) {
      const unsigned short* hc2 = h_bf + (size_t)cur * (B_ * H_);
      const unsigned short* hr  = hc2 + (size_t)(wgid * 16 + lm) * H_;
      const unsigned short* wr  = fcw_bf + (size_t)(wv * 16 + lm) * H_;
      v8f acc = {};
      for (int kc = 0; kc < 16; ++kc) {
        const int ko = kc * 32 + hb * 8;
        acc = wmma_bf16(ld_frag(hr + ko), ld_frag(wr + ko), acc);
      }
      const int i = wv * 16 + lm;
      const float bias = fc_b[i];
#pragma unroll
      for (int vi = 0; vi < 8; ++vi) {
        const int b = wgid * 16 + hb * 8 + vi;
        const float v = acc[vi] + bias;
        out[((size_t)b * HOR + p) * I_ + i] = v;
        seq_bf[((size_t)b * MAXT + (T_ + p)) * I_ + i] = f2bf(v);
      }
    }
    ++ep;
    grid_barrier(bar, ep * NWG);
  }
}

extern "C" void kernel_launch(void* const* d_in, const int* in_sizes, int n_in,
                              void* d_out, int out_size, void* d_ws, size_t ws_size,
                              hipStream_t stream) {
  (void)in_sizes; (void)n_in; (void)out_size; (void)ws_size;
  const float* x   = (const float*)d_in[0];
  const float* wih = (const float*)d_in[1];
  const float* whh = (const float*)d_in[2];
  const float* bih = (const float*)d_in[3];
  const float* bhh = (const float*)d_in[4];
  const float* fcw = (const float*)d_in[5];
  const float* fcb = (const float*)d_in[6];

  char* ws = (char*)d_ws;
  unsigned short* whh_bf = (unsigned short*)(ws + OFF_WHH);
  unsigned short* wih_bf = (unsigned short*)(ws + OFF_WIH);
  unsigned short* fcw_bf = (unsigned short*)(ws + OFF_FCW);
  unsigned short* seq_bf = (unsigned short*)(ws + OFF_SEQ);
  unsigned short* h_bf   = (unsigned short*)(ws + OFF_HBF);
  unsigned*       bar    = (unsigned*)(ws + OFF_BAR);

  gru_init_convert<<<2048, 256, 0, stream>>>(x, wih, whh, fcw, whh_bf, wih_bf,
                                             fcw_bf, seq_bf, h_bf, bar);

  (void)hipFuncSetAttribute((const void*)gru_persistent,
                            hipFuncAttributeMaxDynamicSharedMemorySize, SMEM_BYTES);
  gru_persistent<<<NWG, 256, SMEM_BYTES, stream>>>(bih, bhh, fcb, whh_bf, wih_bf,
                                                   fcw_bf, seq_bf, h_bf,
                                                   (float*)d_out, bar);
}
